// GNN_76184129896787
// MI455X (gfx1250) — matrix operand order repaired and checked by reading, not compiled, not run on hardware
//
#include <hip/hip_runtime.h>
#include <hip/hip_bf16.h>

#define N_NODES 50000
#define N_EDGES 800000
#define ETOT    (N_EDGES + N_NODES)   // with self loops
#define IN_DIM  128
#define HID     32
#define HEADS   8
#define HC      256                    // HEADS*HID
#define N_CLASSES 10
#define N_GRAPHS  64

typedef __attribute__((ext_vector_type(16))) __bf16 v16bf;
typedef __attribute__((ext_vector_type(8)))  float  v8f;

// Float atomic max via ordered-int encoding; location must be initialized with
// 0xFF bytes (a NaN pattern that both signed-max and unsigned-min will replace).
__device__ __forceinline__ void atomicMaxF32(float* addr, float v) {
  int iv = __float_as_int(v);
  if (iv >= 0) atomicMax((int*)addr, iv);
  else         atomicMin((unsigned int*)addr, (unsigned int)iv);
}

// ---------------------------------------------------------------------------
// Pack a row-major f32 weight matrix B [K, Ncols] into per-lane WMMA B-operand
// layout (ISA 7.12.2, 16-bit B 32x16): tile (kt, nt), lane l covers column
// nt*16 + l%16; element t of the lane's v16bf holds K = kt*32 + t + 8*(l/16) + (t>=8 ? 8 : 0).
// One thread per (tile, lane); writes 16 bf16 = 32 contiguous bytes.
// ---------------------------------------------------------------------------
__global__ void k_pack_b(const float* __restrict__ B, __bf16* __restrict__ Bp,
                         int K, int Ncols) {
  int t = blockIdx.x * blockDim.x + threadIdx.x;
  int ktiles = K >> 5;
  int ntiles = ktiles * (Ncols >> 4);
  if (t >= ntiles * 32) return;
  int lane = t & 31, tile = t >> 5;
  int nt = tile / ktiles, kt = tile - nt * ktiles;
  int col = nt * 16 + (lane & 15);
  int kbase = kt * 32 + 8 * (lane >> 4);
  v16bf v;
#pragma unroll
  for (int i = 0; i < 16; ++i)
    v[i] = (__bf16)B[(long long)(kbase + i + ((i & 8) ? 8 : 0)) * Ncols + col];
  *(v16bf*)(Bp + (size_t)t * 16) = v;
}

// ---------------------------------------------------------------------------
// WMMA dual GEMM:  C = act( scale(A1) @ B1 + A2 @ B2 + bias )
// One wave per 16x16 output tile. A [M,K] row-major f32 (converted on load,
// vectorizes to b128 since each lane reads two contiguous 8-float runs);
// B pre-packed bf16 in register layout -> one aligned v16bf load per k-step.
// A-layout (16-bit A 16x32): lane l row=l%16; element t holds
//   K = t + 8*(l/16) + (t>=8 ? 8 : 0).
// C/D: VGPR i -> row i + 8*(l/16), col = l%16.
// ---------------------------------------------------------------------------
__global__ void k_wmma_gemm(const float* __restrict__ A1, const __bf16* __restrict__ B1p, int K1,
                            const float* __restrict__ A2, const __bf16* __restrict__ B2p, int K2,
                            const float* __restrict__ deg,   // if non-null: scale A1 rows by 1/max(deg,1)
                            const float* __restrict__ bias,  // per-column, may be null
                            float* __restrict__ C, int Ncols, int relu) {
  const int lane = threadIdx.x & 31;
  const int mrow = lane & 15;
  const int hi   = lane >> 4;
  const int m0   = blockIdx.x << 4;
  const int n0   = blockIdx.y << 4;
  const int row  = m0 + mrow;
  const int col  = n0 + mrow;

  float scale = 1.0f;
  if (deg) scale = 1.0f / fmaxf(deg[row], 1.0f);

  v8f acc = {};
  {
    const int ktiles = K1 >> 5;
    const float* arow = A1 + (long long)row * K1;
    const v16bf* bp = (const v16bf*)B1p + ((size_t)blockIdx.y * ktiles) * 32 + lane;
    for (int kt = 0; kt < ktiles; ++kt) {
      v16bf a;
      int k0 = kt * 32;
#pragma unroll
      for (int t = 0; t < 16; ++t) {
        int ka = k0 + t + 8 * hi + ((t & 8) ? 8 : 0);
        a[t] = (__bf16)(arow[ka] * scale);
      }
      v16bf b = bp[(size_t)kt * 32];
      acc = __builtin_amdgcn_wmma_f32_16x16x32_bf16(false, a, false, b, (short)0, acc, false, false);
    }
  }
  if (A2) {
    const int ktiles = K2 >> 5;
    const float* arow2 = A2 + (long long)row * K2;
    const v16bf* bp = (const v16bf*)B2p + ((size_t)blockIdx.y * ktiles) * 32 + lane;
    for (int kt = 0; kt < ktiles; ++kt) {
      v16bf a;
      int k0 = kt * 32;
#pragma unroll
      for (int t = 0; t < 16; ++t) {
        int ka = k0 + t + 8 * hi + ((t & 8) ? 8 : 0);
        a[t] = (__bf16)arow2[ka];
      }
      v16bf b = bp[(size_t)kt * 32];
      acc = __builtin_amdgcn_wmma_f32_16x16x32_bf16(false, a, false, b, (short)0, acc, false, false);
    }
  }
  float bv = bias ? bias[col] : 0.0f;
#pragma unroll
  for (int i = 0; i < 8; ++i) {
    float v = acc[i] + bv;
    if (relu) v = fmaxf(v, 0.0f);
    C[(long long)(m0 + i + 8 * hi) * Ncols + col] = v;
  }
}

// a_src[n,h] = <h[n,h,:], att_src[h,:]>, same for dst. One thread per (n, head).
__global__ void k_att(const float* __restrict__ h, const float* __restrict__ att_s,
                      const float* __restrict__ att_d,
                      float* __restrict__ as_, float* __restrict__ ad_) {
  int t = blockIdx.x * blockDim.x + threadIdx.x;
  if (t >= N_NODES * HEADS) return;
  int n = t >> 3, hd = t & 7;
  const float* hp = h + (long long)n * HC + hd * HID;
  const float* ws = att_s + hd * HID;
  const float* wd = att_d + hd * HID;
  float ss = 0.f, sd = 0.f;
#pragma unroll 8
  for (int c = 0; c < HID; ++c) { float v = hp[c]; ss += v * ws[c]; sd += v * wd[c]; }
  as_[t] = ss; ad_[t] = sd;
}

__device__ __forceinline__ void edge_sd(int e, const int* src, const int* dst, int& s, int& d) {
  if (e < N_EDGES) { s = src[e]; d = dst[e]; } else { s = d = e - N_EDGES; }
}

// Pass 1: segment max of leaky-relu logits per (dst, head).
__global__ void k_edge_max(const int* __restrict__ src, const int* __restrict__ dst,
                           const float* __restrict__ as_, const float* __restrict__ ad_,
                           float* __restrict__ m_) {
  int t = blockIdx.x * blockDim.x + threadIdx.x;
  if (t >= ETOT * HEADS) return;
  int e = t >> 3, hd = t & 7, s, d;
  edge_sd(e, src, dst, s, d);
  float lg = as_[s * HEADS + hd] + ad_[d * HEADS + hd];
  lg = (lg > 0.f) ? lg : 0.2f * lg;
  atomicMaxF32(&m_[d * HEADS + hd], lg);
}

// Pass 2: z[dst,h] += exp(logit - m).
__global__ void k_edge_expsum(const int* __restrict__ src, const int* __restrict__ dst,
                              const float* __restrict__ as_, const float* __restrict__ ad_,
                              const float* __restrict__ m_, float* __restrict__ z_) {
  int t = blockIdx.x * blockDim.x + threadIdx.x;
  if (t >= ETOT * HEADS) return;
  int e = t >> 3, hd = t & 7, s, d;
  edge_sd(e, src, dst, s, d);
  float lg = as_[s * HEADS + hd] + ad_[d * HEADS + hd];
  lg = (lg > 0.f) ? lg : 0.2f * lg;
  atomicAdd(&z_[d * HEADS + hd], __expf(lg - m_[d * HEADS + hd]));
}

// Pass 3: out[dst, h*32+c] += alpha * h[src, h*32+c]. One wave per edge,
// lane covers channel 32*i + lane for head i (i = 0..7).
__global__ void k_edge_msg(const int* __restrict__ src, const int* __restrict__ dst,
                           const float* __restrict__ as_, const float* __restrict__ ad_,
                           const float* __restrict__ m_, const float* __restrict__ z_,
                           const float* __restrict__ h, float* __restrict__ out) {
  int t = blockIdx.x * blockDim.x + threadIdx.x;
  int e = t >> 5, lane = t & 31;
  if (e >= ETOT) return;
  int s, d;
  edge_sd(e, src, dst, s, d);
  const float* hs = h + (long long)s * HC;
  float* od = out + (long long)d * HC;
#pragma unroll
  for (int i = 0; i < HEADS; ++i) {
    float lg = as_[s * HEADS + i] + ad_[d * HEADS + i];
    lg = (lg > 0.f) ? lg : 0.2f * lg;
    float alpha = __expf(lg - m_[d * HEADS + i]) / (z_[d * HEADS + i] + 1e-16f);
    int c = (i << 5) + lane;
    atomicAdd(&od[c], hs[c] * alpha);
  }
}

__global__ void k_bias_relu(float* __restrict__ x1, const float* __restrict__ bias) {
  long long t = (long long)blockIdx.x * blockDim.x + threadIdx.x;
  if (t >= (long long)N_NODES * HC) return;
  x1[t] = fmaxf(x1[t] + bias[t & (HC - 1)], 0.f);
}

__global__ void k_deg(const int* __restrict__ dst, float* __restrict__ deg) {
  int e = blockIdx.x * blockDim.x + threadIdx.x;
  if (e >= N_EDGES) return;
  atomicAdd(&deg[dst[e]], 1.0f);
}

// out[dst,:] += X[src,:]; wave per edge, lane strides channels.
__global__ void k_scatter_add(const float* __restrict__ X, const int* __restrict__ src,
                              const int* __restrict__ dst, float* __restrict__ out, int D) {
  int t = blockIdx.x * blockDim.x + threadIdx.x;
  int e = t >> 5, lane = t & 31;
  if (e >= N_EDGES) return;
  int s = src[e], d = dst[e];
  for (int c = lane; c < D; c += 32)
    atomicAdd(&out[(long long)d * D + c], X[(long long)s * D + c]);
}

// Global mean pool accumulation: wave per node, lane per feature.
__global__ void k_pool(const float* __restrict__ x4, const int* __restrict__ batch,
                       float* __restrict__ gsum, float* __restrict__ gcnt) {
  int t = blockIdx.x * blockDim.x + threadIdx.x;
  int n = t >> 5, lane = t & 31;
  if (n >= N_NODES) return;
  int b = batch[n];
  atomicAdd(&gsum[b * HID + lane], x4[(long long)n * HID + lane]);
  if (lane == 0) atomicAdd(&gcnt[b], 1.0f);
}

// MLP head: one thread per graph (64 threads total; trivial FLOPs).
__global__ void k_head(const float* __restrict__ gsum, const float* __restrict__ gcnt,
                       const float* __restrict__ fc1W, const float* __restrict__ fc1b,
                       const float* __restrict__ fc2W, const float* __restrict__ fc2b,
                       float* __restrict__ out) {
  int g = threadIdx.x;
  if (g >= N_GRAPHS) return;
  float mean[HID], hid1[HID];
  float inv = 1.0f / fmaxf(gcnt[g], 1.0f);
  for (int k = 0; k < HID; ++k) mean[k] = gsum[g * HID + k] * inv;
  for (int j = 0; j < HID; ++j) {
    float v = fc1b[j];
    for (int k = 0; k < HID; ++k) v += mean[k] * fc1W[k * HID + j];
    hid1[j] = fmaxf(v, 0.f);
  }
  for (int c = 0; c < N_CLASSES; ++c) {
    float v = fc2b[c];
    for (int j = 0; j < HID; ++j) v += hid1[j] * fc2W[j * N_CLASSES + c];
    out[g * N_CLASSES + c] = v;
  }
}

extern "C" void kernel_launch(void* const* d_in, const int* in_sizes, int n_in,
                              void* d_out, int out_size, void* d_ws, size_t ws_size,
                              hipStream_t stream) {
  const float* x        = (const float*)d_in[0];
  const int*   ei       = (const int*)d_in[1];
  const int*   src      = ei;
  const int*   dst      = ei + N_EDGES;
  const int*   batch    = (const int*)d_in[2];
  const float* gat_W    = (const float*)d_in[3];
  const float* att_src  = (const float*)d_in[4];
  const float* att_dst  = (const float*)d_in[5];
  const float* gat_b    = (const float*)d_in[6];
  const float* W2_rel   = (const float*)d_in[7];
  const float* b2       = (const float*)d_in[8];
  const float* W2_root  = (const float*)d_in[9];
  const float* W3_l     = (const float*)d_in[10];
  const float* b3       = (const float*)d_in[11];
  const float* W3_r     = (const float*)d_in[12];
  const float* W4_rel   = (const float*)d_in[13];
  const float* b4       = (const float*)d_in[14];
  const float* W4_root  = (const float*)d_in[15];
  const float* fc1_W    = (const float*)d_in[16];
  const float* fc1_b    = (const float*)d_in[17];
  const float* fc2_W    = (const float*)d_in[18];
  const float* fc2_b    = (const float*)d_in[19];

  // Workspace layout (floats; all region sizes are multiples of 8 floats so
  // the trailing bf16 packed-weight region stays 32-byte aligned for v16bf).
  float*  ws  = (float*)d_ws;
  size_t  off = 0;
  float* h    = ws + off; off += (size_t)N_NODES * HC;     // h / agg (reused)
  float* x1   = ws + off; off += (size_t)N_NODES * HC;
  float* as_  = ws + off; off += (size_t)N_NODES * HEADS;
  float* ad_  = ws + off; off += (size_t)N_NODES * HEADS;
  float* m_   = ws + off; off += (size_t)N_NODES * HEADS;
  float* z_   = ws + off; off += (size_t)N_NODES * HEADS;
  float* x2   = ws + off; off += (size_t)N_NODES * HID;
  float* x3   = ws + off; off += (size_t)N_NODES * HID;
  float* x4   = ws + off; off += (size_t)N_NODES * HID;
  float* deg  = ws + off; off += (size_t)N_NODES;
  float* gsum = ws + off; off += (size_t)N_GRAPHS * HID;
  float* gcnt = ws + off; off += (size_t)N_GRAPHS;

  // Packed bf16 weights (WMMA register layout). Sizes in bf16 elements; each
  // is a multiple of 512 so sub-regions stay 32B-aligned.
  __bf16* wb = (__bf16*)(ws + off);
  size_t boff = 0;
  const size_t SZ_GATW = (IN_DIM / 32) * (HC / 16) * 512;   // 32768
  const size_t SZ_W2   = (HC / 32) * (HID / 16) * 512;      // 8192
  const size_t SZ_W34  = (HID / 32) * (HID / 16) * 512;     // 1024
  __bf16* p_gatW  = wb + boff; boff += SZ_GATW;
  __bf16* p_W2rel = wb + boff; boff += SZ_W2;
  __bf16* p_W2rt  = wb + boff; boff += SZ_W2;
  __bf16* p_W3l   = wb + boff; boff += SZ_W34;
  __bf16* p_W3r   = wb + boff; boff += SZ_W34;
  __bf16* p_W4rel = wb + boff; boff += SZ_W34;
  __bf16* p_W4rt  = wb + boff; boff += SZ_W34;
  off += (boff + 1) / 2;
  if (ws_size < off * sizeof(float)) return;

  float* agg = h;  // alias, used after GAT layer

  const int B = 256;
  dim3 blk(B);

  // ---- zero / init accumulators (graph-capture-safe async memsets) ----
  hipMemsetAsync(m_,   0xFF, (size_t)N_NODES * HEADS * 4, stream);  // -inf-like for atomicMaxF32
  hipMemsetAsync(z_,   0,    (size_t)N_NODES * HEADS * 4, stream);
  hipMemsetAsync(x1,   0,    (size_t)N_NODES * HC * 4,    stream);
  hipMemsetAsync(deg,  0,    (size_t)N_NODES * 4,         stream);
  hipMemsetAsync(gsum, 0,    (size_t)(N_GRAPHS * HID + N_GRAPHS) * 4, stream);  // gsum+gcnt contiguous

  // ---- pack weights to WMMA bf16 layout (tiny, recomputed each call) ----
  k_pack_b<<<(int)((SZ_GATW / 16 + B - 1) / B), blk, 0, stream>>>(gat_W, p_gatW, IN_DIM, HC);
  k_pack_b<<<(int)((SZ_W2 / 16 + B - 1) / B),   blk, 0, stream>>>(W2_rel,  p_W2rel, HC,  HID);
  k_pack_b<<<(int)((SZ_W2 / 16 + B - 1) / B),   blk, 0, stream>>>(W2_root, p_W2rt,  HC,  HID);
  k_pack_b<<<(int)((SZ_W34 / 16 + B - 1) / B),  blk, 0, stream>>>(W3_l,    p_W3l,   HID, HID);
  k_pack_b<<<(int)((SZ_W34 / 16 + B - 1) / B),  blk, 0, stream>>>(W3_r,    p_W3r,   HID, HID);
  k_pack_b<<<(int)((SZ_W34 / 16 + B - 1) / B),  blk, 0, stream>>>(W4_rel,  p_W4rel, HID, HID);
  k_pack_b<<<(int)((SZ_W34 / 16 + B - 1) / B),  blk, 0, stream>>>(W4_root, p_W4rt,  HID, HID);

  // ---- GAT: h = x @ gat_W   [50000,128]x[128,256]  (WMMA bf16) ----
  k_wmma_gemm<<<dim3(N_NODES / 16, HC / 16), dim3(32), 0, stream>>>(
      x, p_gatW, IN_DIM, nullptr, nullptr, 0, nullptr, nullptr, h, HC, 0);

  k_att<<<(N_NODES * HEADS + B - 1) / B, blk, 0, stream>>>(h, att_src, att_dst, as_, ad_);

  int eh_blocks = (ETOT * HEADS + B - 1) / B;
  k_edge_max<<<eh_blocks, blk, 0, stream>>>(src, dst, as_, ad_, m_);
  k_edge_expsum<<<eh_blocks, blk, 0, stream>>>(src, dst, as_, ad_, m_, z_);
  k_edge_msg<<<((size_t)ETOT * 32 + B - 1) / B, blk, 0, stream>>>(src, dst, as_, ad_, m_, z_, h, x1);
  k_bias_relu<<<((size_t)N_NODES * HC + B - 1) / B, blk, 0, stream>>>(x1, gat_b);

  k_deg<<<(N_EDGES + B - 1) / B, blk, 0, stream>>>(dst, deg);

  // ---- GraphConv 2: x2 = relu(sum_agg(x1) @ W2_rel + x1 @ W2_root + b2) ----
  hipMemsetAsync(agg, 0, (size_t)N_NODES * HC * 4, stream);
  k_scatter_add<<<((size_t)N_EDGES * 32 + B - 1) / B, blk, 0, stream>>>(x1, src, dst, agg, HC);
  k_wmma_gemm<<<dim3(N_NODES / 16, HID / 16), dim3(32), 0, stream>>>(
      agg, p_W2rel, HC, x1, p_W2rt, HC, nullptr, b2, x2, HID, 1);

  // ---- SAGE 3: x3 = relu(mean_agg(x2) @ W3_l + x2 @ W3_r + b3) ----
  hipMemsetAsync(agg, 0, (size_t)N_NODES * HID * 4, stream);
  k_scatter_add<<<((size_t)N_EDGES * 32 + B - 1) / B, blk, 0, stream>>>(x2, src, dst, agg, HID);
  k_wmma_gemm<<<dim3(N_NODES / 16, HID / 16), dim3(32), 0, stream>>>(
      agg, p_W3l, HID, x2, p_W3r, HID, deg, b3, x3, HID, 1);

  // ---- GraphConv 4: x4 = relu(sum_agg(x3) @ W4_rel + x3 @ W4_root + b4) ----
  hipMemsetAsync(agg, 0, (size_t)N_NODES * HID * 4, stream);
  k_scatter_add<<<((size_t)N_EDGES * 32 + B - 1) / B, blk, 0, stream>>>(x3, src, dst, agg, HID);
  k_wmma_gemm<<<dim3(N_NODES / 16, HID / 16), dim3(32), 0, stream>>>(
      agg, p_W4rel, HID, x3, p_W4rt, HID, nullptr, b4, x4, HID, 1);

  // ---- global mean pool + MLP head ----
  k_pool<<<((size_t)N_NODES * 32 + B - 1) / B, blk, 0, stream>>>(x4, batch, gsum, gcnt);
  k_head<<<1, dim3(64), 0, stream>>>(gsum, gcnt, fc1_W, fc1_b, fc2_W, fc2_b, (float*)d_out);
}
